// SupeConMemoryLoss_67869073212268
// MI455X (gfx1250) — matrix operand order repaired and checked by reading, not compiled, not run on hardware
//
#include <hip/hip_runtime.h>

typedef __attribute__((ext_vector_type(16))) __bf16 v16bf;
typedef __attribute__((ext_vector_type(8)))  __bf16 v8bf;
typedef __attribute__((ext_vector_type(8)))  float  v8f;

#define INV_T (1.0f / 0.07f)

// ---------- shared-memory block reductions (256 threads = 8 wave32) ----------
__device__ inline float blk_sum(float v, float* red) {
  __syncthreads();
  #pragma unroll
  for (int o = 16; o; o >>= 1) v += __shfl_xor(v, o, 32);
  if ((threadIdx.x & 31) == 0) red[threadIdx.x >> 5] = v;
  __syncthreads();
  float t = 0.f;
  if (threadIdx.x == 0) {
    for (int i = 0; i < (int)(blockDim.x >> 5); ++i) t += red[i];
  }
  return t;  // valid on thread 0 only
}

__device__ inline float blk_max_bcast(float v, float* red) {
  __syncthreads();
  #pragma unroll
  for (int o = 16; o; o >>= 1) v = fmaxf(v, __shfl_xor(v, o, 32));
  if ((threadIdx.x & 31) == 0) red[threadIdx.x >> 5] = v;
  __syncthreads();
  if (threadIdx.x == 0) {
    float t = red[0];
    for (int i = 1; i < (int)(blockDim.x >> 5); ++i) t = fmaxf(t, red[i]);
    red[0] = t;
  }
  __syncthreads();
  return red[0];  // broadcast to all threads
}

// ---------- kernel 1: row-normalize q, write bf16 hi/lo split ----------
__global__ __launch_bounds__(256) void k_normalize(const float* __restrict__ q,
                                                   __bf16* __restrict__ qhi,
                                                   __bf16* __restrict__ qlo,
                                                   int D) {
  __shared__ float red[8];
  __shared__ float s_inv;
  const int row = blockIdx.x;
  const float* qr = q + (size_t)row * D;
  float ss = 0.f;
  for (int j = threadIdx.x; j < D; j += blockDim.x) {
    float v = qr[j];
    ss += v * v;
  }
  float tot = blk_sum(ss, red);
  if (threadIdx.x == 0) s_inv = 1.0f / fmaxf(sqrtf(tot), 1e-8f);
  __syncthreads();
  const float inv = s_inv;
  for (int j = threadIdx.x; j < D; j += blockDim.x) {
    float x = qr[j] * inv;
    __bf16 h = (__bf16)x;
    float  hf = (float)h;
    qhi[(size_t)row * D + j] = h;
    qlo[(size_t)row * D + j] = (__bf16)(x - hf);
  }
}

// ---------- kernel 2: memory-branch row reductions (streams 256 MB) ----------
__global__ __launch_bounds__(256) void k_mem(const float* __restrict__ ml,
                                             const float* __restrict__ mt,
                                             float* __restrict__ sE,
                                             float* __restrict__ sT,
                                             float* __restrict__ sTL,
                                             int M) {
  __shared__ float red[8];
  const int i = blockIdx.x;
  const float4* mlp = (const float4*)(ml + (size_t)i * M);
  const float4* mtp = (const float4*)(mt + (size_t)i * M);
  float se = 0.f, st = 0.f, stl = 0.f;
  for (int s = threadIdx.x; s < (M >> 2); s += blockDim.x) {
    float4 a = mlp[s];
    float4 b = mtp[s];
    se  += __expf(a.x) + __expf(a.y) + __expf(a.z) + __expf(a.w);
    st  += b.x + b.y + b.z + b.w;
    stl += b.x * a.x + b.y * a.y + b.z * a.z + b.w * a.w;
  }
  float r0 = blk_sum(se, red);
  float r1 = blk_sum(st, red);
  float r2 = blk_sum(stl, red);
  if (threadIdx.x == 0) { sE[i] = r0; sT[i] = r1; sTL[i] = r2; }
}

// ---------- WMMA fragment loaders (layouts per CDNA5 ISA) ----------
// A 16x32 bf16: lane L -> row M = L%16; K chunks at (L/16)*8 and 16+(L/16)*8.
__device__ inline v16bf load_afrag(const __bf16* __restrict__ base, int D,
                                   int rowTile, int k, int lane) {
  const int m = rowTile + (lane & 15);
  const int h = lane >> 4;
  const __bf16* p = base + (size_t)m * D + k + h * 8;
  v8bf a0 = *reinterpret_cast<const v8bf*>(p);
  v8bf a1 = *reinterpret_cast<const v8bf*>(p + 16);
  v16bf r;
  #pragma unroll
  for (int i = 0; i < 8; ++i) { r[i] = a0[i]; r[i + 8] = a1[i]; }
  return r;
}

// B 32x16 bf16 (= qn^T tile): lane L -> col N = L%16; K chunks at (L/16)*8 and 16+(L/16)*8.
__device__ inline v16bf load_bfrag(const __bf16* __restrict__ base, int D,
                                   int colTile, int k, int lane) {
  const int n = colTile + (lane & 15);
  const int h = lane >> 4;
  const __bf16* p = base + (size_t)n * D + k + h * 8;
  v8bf b0 = *reinterpret_cast<const v8bf*>(p);
  v8bf b1 = *reinterpret_cast<const v8bf*>(p + 16);
  v16bf r;
  #pragma unroll
  for (int i = 0; i < 8; ++i) { r[i] = b0[i]; r[i + 8] = b1[i]; }
  return r;
}

// ---------- kernel 3: S = qn @ qn^T via bf16x3 split WMMA, f32 accumulate ----------
// Block: 256 threads = 8 waves arranged 4(M) x 2(N); wave tile 32x32 -> block 128x64.
__global__ __launch_bounds__(256) void k_gemm(const __bf16* __restrict__ qhi,
                                              const __bf16* __restrict__ qlo,
                                              float* __restrict__ S,
                                              int N, int D) {
  const int lane = threadIdx.x & 31;
  const int wave = threadIdx.x >> 5;
  const int wm = wave & 3;
  const int wn = wave >> 2;
  const int row0 = blockIdx.y * 128 + wm * 32;
  const int col0 = blockIdx.x * 64 + wn * 32;

  v8f acc[2][2];
  #pragma unroll
  for (int a = 0; a < 2; ++a)
    #pragma unroll
    for (int b = 0; b < 2; ++b)
      acc[a][b] = (v8f)(0.0f);

  for (int k = 0; k < D; k += 32) {
    v16bf aH[2], aL[2], bH[2], bL[2];
    #pragma unroll
    for (int t = 0; t < 2; ++t) {
      aH[t] = load_afrag(qhi, D, row0 + t * 16, k, lane);
      aL[t] = load_afrag(qlo, D, row0 + t * 16, k, lane);
      bH[t] = load_bfrag(qhi, D, col0 + t * 16, k, lane);
      bL[t] = load_bfrag(qlo, D, col0 + t * 16, k, lane);
    }
    #pragma unroll
    for (int mi = 0; mi < 2; ++mi) {
      #pragma unroll
      for (int ni = 0; ni < 2; ++ni) {
        // bf16x3: hi*hi + hi*lo + lo*hi (lo*lo ~ 2^-16 of hi*hi, dropped)
        acc[mi][ni] = __builtin_amdgcn_wmma_f32_16x16x32_bf16(
            false, aH[mi], false, bH[ni], (short)0, acc[mi][ni], false, false);
        acc[mi][ni] = __builtin_amdgcn_wmma_f32_16x16x32_bf16(
            false, aH[mi], false, bL[ni], (short)0, acc[mi][ni], false, false);
        acc[mi][ni] = __builtin_amdgcn_wmma_f32_16x16x32_bf16(
            false, aL[mi], false, bH[ni], (short)0, acc[mi][ni], false, false);
      }
    }
  }

  // C layout: VGPR r, lanes0-15: M=r, N=lane; lanes16-31: M=r+8, N=lane-16.
  #pragma unroll
  for (int mi = 0; mi < 2; ++mi) {
    #pragma unroll
    for (int ni = 0; ni < 2; ++ni) {
      const int cc = col0 + ni * 16 + (lane & 15);
      #pragma unroll
      for (int r = 0; r < 8; ++r) {
        const int rr = row0 + mi * 16 + r + (lane >> 4) * 8;
        S[(size_t)rr * N + cc] = acc[mi][ni][r];
      }
    }
  }
}

// ---------- kernel 4: per-row softmax stats over S (reads S from L2) ----------
__global__ __launch_bounds__(256) void k_src(const float* __restrict__ S,
                                             const int* __restrict__ labels,
                                             float* __restrict__ sumexp,
                                             float* __restrict__ sumposx,
                                             float* __restrict__ cntpos,
                                             int N) {
  __shared__ float red[8];
  const int i = blockIdx.x;
  const float* Si = S + (size_t)i * N;
  const int li = labels[i];
  const int iters = N >> 8;  // N/256 (= 16 for N=4096)
  float x[16];
  float mx = -3.4e38f;
  for (int s = 0; s < iters; ++s) {
    const int j = s * 256 + threadIdx.x;
    float v = Si[j] * INV_T;
    x[s] = v;
    mx = fmaxf(mx, v);
  }
  const float m = blk_max_bcast(mx, red);  // includes diagonal (as in reference)
  float se = 0.f, sp = 0.f, cnt = 0.f;
  for (int s = 0; s < iters; ++s) {
    const int j = s * 256 + threadIdx.x;
    if (j != i) {  // self excluded from exp-sum and positives
      const float xs = x[s] - m;
      se += __expf(xs);
      if (labels[j] == li) { sp += xs; cnt += 1.f; }
    }
  }
  float r0 = blk_sum(se, red);
  float r1 = blk_sum(sp, red);
  float r2 = blk_sum(cnt, red);
  if (threadIdx.x == 0) { sumexp[i] = r0; sumposx[i] = r1; cntpos[i] = r2; }
}

// ---------- kernel 5: final scalar reduce ----------
__global__ __launch_bounds__(256) void k_final(const float* __restrict__ sE,
                                               const float* __restrict__ sT,
                                               const float* __restrict__ sTL,
                                               const float* __restrict__ sumexp,
                                               const float* __restrict__ sumposx,
                                               const float* __restrict__ cntpos,
                                               float* __restrict__ out, int N) {
  __shared__ float red[8];
  float acc = 0.f;
  for (int i = threadIdx.x; i < N; i += blockDim.x) {
    const float denom = sumexp[i] + sE[i];
    const float ld = __logf(denom);
    const float npos = cntpos[i] + sT[i];
    const float num = sumposx[i] + sTL[i] - npos * ld;
    acc += num / npos;
  }
  float tot = blk_sum(acc, red);
  if (threadIdx.x == 0) out[0] = -tot / (float)N;
}

// ---------- launch ----------
extern "C" void kernel_launch(void* const* d_in, const int* in_sizes, int n_in,
                              void* d_out, int out_size, void* d_ws, size_t ws_size,
                              hipStream_t stream) {
  const float* q      = (const float*)d_in[0];
  const int*   labels = (const int*)d_in[1];
  const float* ml     = (const float*)d_in[2];
  const float* mt     = (const float*)d_in[3];

  const int N = in_sizes[1];
  const int D = in_sizes[0] / N;
  const int M = in_sizes[2] / N;

  auto align256 = [](size_t x) { return (x + 255) & ~(size_t)255; };
  char* ws = (char*)d_ws;
  size_t off = 0;
  __bf16* qhi = (__bf16*)(ws + off); off = align256(off + (size_t)N * D * sizeof(__bf16));
  __bf16* qlo = (__bf16*)(ws + off); off = align256(off + (size_t)N * D * sizeof(__bf16));
  float*  S   = (float*)(ws + off);  off = align256(off + (size_t)N * N * sizeof(float));
  float* sE      = (float*)(ws + off); off = align256(off + (size_t)N * sizeof(float));
  float* sT      = (float*)(ws + off); off = align256(off + (size_t)N * sizeof(float));
  float* sTL     = (float*)(ws + off); off = align256(off + (size_t)N * sizeof(float));
  float* sumexp  = (float*)(ws + off); off = align256(off + (size_t)N * sizeof(float));
  float* sumposx = (float*)(ws + off); off = align256(off + (size_t)N * sizeof(float));
  float* cntpos  = (float*)(ws + off); off = align256(off + (size_t)N * sizeof(float));
  (void)ws_size; (void)n_in; (void)out_size;

  k_normalize<<<N, 256, 0, stream>>>(q, qhi, qlo, D);
  k_mem<<<N, 256, 0, stream>>>(ml, mt, sE, sT, sTL, M);
  dim3 ggrid(N / 64, N / 128);
  k_gemm<<<ggrid, 256, 0, stream>>>(qhi, qlo, S, N, D);
  k_src<<<N, 256, 0, stream>>>(S, labels, sumexp, sumposx, cntpos, N);
  k_final<<<1, 256, 0, stream>>>(sE, sT, sTL, sumexp, sumposx, cntpos,
                                 (float*)d_out, N);
}
